// StateSpaceGRU_57956288692497
// MI455X (gfx1250) — compile-verified, matrix operand order checked by reading, not run on hardware
//
#include <hip/hip_runtime.h>
#include <hip/hip_bf16.h>

// ---------------------------------------------------------------------------
// StateSpaceGRU for MI455X (gfx1250), wave32 + WMMA bf16 + cluster barriers
// + Tensor Data Mover staging of recurrent weights into LDS.
//
// B=64,T=512,NU=256,NX=1024.  4 batch-groups of 16 rows (WMMA M=16); each
// group's scan runs on a 16-workgroup cluster slice; NX=1024 split as
// 16 WGs x 4 waves x 16 columns.  Two sync points per timestep.
// Uz/Uf slices live in LDS (256 KB dynamic, TDM-staged); Ur streams from L2.
// ---------------------------------------------------------------------------

constexpr int kB  = 64;
constexpr int kT  = 512;
constexpr int kNU = 256;
constexpr int kNX = 1024;

constexpr int kWaves    = 4;     // waves per workgroup
constexpr int kWGsPerGp = 16;    // workgroups per batch-group (== cluster size)
constexpr int kGroups   = 4;     // batch groups of 16 rows
constexpr int kKTu      = kNX / 32;  // 32 K-tiles for recurrent GEMMs
constexpr int kKTw      = kNU / 32;  // 8  K-tiles for input-proj GEMMs
constexpr int kNT       = kNX / 16;  // 64 N-tiles

// ---- workspace layout (bytes) ---------------------------------------------
constexpr size_t OFF_UBF = 0;                                   // u in bf16
constexpr size_t SZ_UBF  = (size_t)kB * kT * kNU * 2;           // 16 MB
constexpr size_t OFF_WPK = OFF_UBF + SZ_UBF;                    // packed W (3)
constexpr size_t SZ_WPK  = (size_t)3 * kNT * kKTw * 32 * 16 * 2;// 1.5 MB
constexpr size_t OFF_UPK = OFF_WPK + SZ_WPK;                    // packed U (3)
constexpr size_t SZ_UPK  = (size_t)3 * kNT * kKTu * 32 * 16 * 2;// 6 MB
constexpr size_t OFF_XF  = OFF_UPK + SZ_UPK;                    // state fp32
constexpr size_t SZ_XF   = (size_t)kB * kNX * 4;
constexpr size_t OFF_XBF = OFF_XF + SZ_XF;                      // state bf16
constexpr size_t SZ_XBF  = (size_t)kB * kNX * 2;
constexpr size_t OFF_GBF = OFF_XBF + SZ_XBF;                    // g = f*x bf16
constexpr size_t SZ_GBF  = (size_t)kB * kNX * 2;
constexpr size_t OFF_BAR = OFF_GBF + SZ_GBF;                    // barrier state
constexpr size_t SZ_BAR  = 512;

typedef __attribute__((ext_vector_type(16))) __bf16 v16bf;
typedef __attribute__((ext_vector_type(8)))  __bf16 v8bf;
typedef __attribute__((ext_vector_type(8)))  float  v8f;
typedef __attribute__((ext_vector_type(4)))  unsigned int v4u;
typedef __attribute__((ext_vector_type(8)))  int v8i;
typedef __attribute__((ext_vector_type(4)))  int v4i;
typedef __attribute__((address_space(3)))    unsigned short* lds_us_ptr;

// ---- helpers ---------------------------------------------------------------
__device__ __forceinline__ unsigned short f2bf(float f) {
  unsigned u = __float_as_uint(f);
  unsigned r = (u + 0x7FFFu + ((u >> 16) & 1u)) >> 16;  // round-nearest-even
  return (unsigned short)r;
}

// A operand: row-major source, lane holds k0+8*hi..+7 and k0+16+8*hi..+7.
__device__ __forceinline__ v16bf ld_a(const unsigned short* p) {
  v8bf lo = *(const v8bf*)p;
  v8bf hh = *(const v8bf*)(p + 16);
  return __builtin_shufflevector(lo, hh, 0,1,2,3,4,5,6,7,8,9,10,11,12,13,14,15);
}
// B operand: pre-packed, 16 contiguous bf16 per lane.
__device__ __forceinline__ v16bf ld_b(const unsigned short* p) {
  v8bf lo = *(const v8bf*)p;
  v8bf hh = *(const v8bf*)(p + 8);
  return __builtin_shufflevector(lo, hh, 0,1,2,3,4,5,6,7,8,9,10,11,12,13,14,15);
}

__device__ __forceinline__ v8f wmma_bf16(v16bf a, v16bf b, v8f c) {
  return __builtin_amdgcn_wmma_f32_16x16x32_bf16(false, a, false, b,
                                                 (short)0, c, false, false);
}

__device__ __forceinline__ float fsigmoid(float x) {
  return 1.0f / (1.0f + __expf(-x));
}
__device__ __forceinline__ float ftanh(float x) {
  return 2.0f / (1.0f + __expf(-2.0f * x)) - 1.0f;  // saturates to +-1
}

// ---- Tensor Data Mover: 1D contiguous global->LDS copy ---------------------
// D# per CDNA5 ISA.  n_dwords <= 65535.
// group0: [1:0]=count=1, [63:32]=lds_addr, [120:64]=global_addr, [127:126]=2.
// group1: [17:16]=data_size(2 => 4B), tensor_dim0@[79:48], tensor_dim1@[111:80]=1,
//         tile_dim0@[127:112], tensor_dim0_stride@[207:160].
__device__ __forceinline__ void tdm_load_1d(unsigned lds_byte_off,
                                            const void* gptr, unsigned n_dwords) {
  unsigned long long ga = (unsigned long long)gptr;
  v4u g0;
  g0[0] = 1u;                                           // count=1 (user D#)
  g0[1] = lds_byte_off;                                 // LDS address (bytes)
  g0[2] = (unsigned)(ga & 0xffffffffu);                 // global_addr[31:0]
  g0[3] = (unsigned)((ga >> 32) & 0x01ffffffu) | (2u << 30);  // addr[56:32]|type=2
  v8i g1;
  g1[0] = (int)(2u << 16);                              // data_size = 4 bytes
  g1[1] = (int)((n_dwords & 0xffffu) << 16);            // tensor_dim0[15:0]
  g1[2] = (int)(((n_dwords >> 16) & 0xffffu) | (1u << 16)); // td0[31:16], td1=1
  g1[3] = (int)((n_dwords & 0xffffu) << 16);            // tile_dim0
  g1[4] = 0;                                            // tile_dim1/2 unused
  g1[5] = (int)n_dwords;                                // tensor_dim0_stride
  g1[6] = 0;
  g1[7] = 0;
  v4i z4 = {0, 0, 0, 0};
#if __clang_major__ >= 23
  v8i z8 = {0, 0, 0, 0, 0, 0, 0, 0};
  __builtin_amdgcn_tensor_load_to_lds(g0, g1, z4, z4, z8, 0);
#else
  __builtin_amdgcn_tensor_load_to_lds(g0, g1, z4, z4, 0);
#endif
}

// ---- prep kernels ----------------------------------------------------------
__global__ void k_convert_u(const float* __restrict__ u,
                            unsigned short* __restrict__ ubf, int n) {
  int i = blockIdx.x * blockDim.x + threadIdx.x;
  if (i < n) ubf[i] = f2bf(u[i]);
}

// Pack a (K x N) fp32 matrix into WMMA bf16 B-operand tiles:
// [gate][ntile][ktile][lane][16].  Element j (=2q+p) of a lane maps to
// k = kt*32 + 16*(q>>2) + 8*(lane>>4) + 2*(q&3) + p ;  n = nt*16 + (lane&15).
__global__ void k_pack_w(const float* __restrict__ Wz, const float* __restrict__ Wf,
                         const float* __restrict__ Wr, unsigned short* __restrict__ dst) {
  int idx = blockIdx.x * blockDim.x + threadIdx.x;   // 3*64*8*32 threads
  if (idx >= 3 * kNT * kKTw * 32) return;
  int lane = idx & 31, kt = (idx >> 5) & 7, ntl = (idx >> 8) & 63, g = idx >> 14;
  const float* W = (g == 0) ? Wz : (g == 1) ? Wf : Wr;
  int hi = lane >> 4, n = ntl * 16 + (lane & 15);
  unsigned short* d = dst + (size_t)idx * 16;
  for (int j = 0; j < 16; ++j) {
    int q = j >> 1, p = j & 1;
    int k = kt * 32 + ((q >> 2) << 4) + (hi << 3) + ((q & 3) << 1) + p;
    d[j] = f2bf(W[(size_t)k * kNX + n]);
  }
}

__global__ void k_pack_u(const float* __restrict__ Uz, const float* __restrict__ Uf,
                         const float* __restrict__ Ur, unsigned short* __restrict__ dst) {
  int idx = blockIdx.x * blockDim.x + threadIdx.x;   // 3*64*32*32 threads
  if (idx >= 3 * kNT * kKTu * 32) return;
  int lane = idx & 31, kt = (idx >> 5) & 31, ntl = (idx >> 10) & 63, g = idx >> 16;
  const float* U = (g == 0) ? Uz : (g == 1) ? Uf : Ur;
  int hi = lane >> 4, n = ntl * 16 + (lane & 15);
  unsigned short* d = dst + (size_t)idx * 16;
  for (int j = 0; j < 16; ++j) {
    int q = j >> 1, p = j & 1;
    int k = kt * 32 + ((q >> 2) << 4) + (hi << 3) + ((q & 3) << 1) + p;
    d[j] = f2bf(U[(size_t)k * kNX + n]);
  }
}

__global__ void k_init_state(const float* __restrict__ x0, float* __restrict__ xf,
                             unsigned short* __restrict__ xbf, unsigned* __restrict__ bar) {
  int i = blockIdx.x * blockDim.x + threadIdx.x;
  if (i < kB * kNX) {
    float v = x0[i];
    xf[i] = v;
    xbf[i] = f2bf(v);
  }
  if (i < 128) bar[i] = 0u;   // zero barrier counters/generations each launch
}

// ---- cluster/group sync ----------------------------------------------------
// HW cluster barrier (ISA protocol: one wave signals, all wait; NOP when not
// dispatched as a cluster) + atomic sense barrier fallback for correctness.
__device__ __forceinline__ void cluster_sync(unsigned* cnt, unsigned* gen,
                                             unsigned& mygen, int tid) {
  __builtin_amdgcn_fence(__ATOMIC_RELEASE, "agent");   // flush this wave's stores
  __syncthreads();
  mygen++;
  if (tid == 0) {
    asm volatile("s_barrier_signal -3" ::: "memory");  // cluster barrier arrive
    unsigned prev = __hip_atomic_fetch_add(cnt, 1u, __ATOMIC_ACQ_REL,
                                           __HIP_MEMORY_SCOPE_AGENT);
    if (prev == (unsigned)(kWGsPerGp - 1)) {
      __hip_atomic_store(cnt, 0u, __ATOMIC_RELAXED, __HIP_MEMORY_SCOPE_AGENT);
      __hip_atomic_fetch_add(gen, 1u, __ATOMIC_RELEASE, __HIP_MEMORY_SCOPE_AGENT);
    }
    while (__hip_atomic_load(gen, __ATOMIC_ACQUIRE, __HIP_MEMORY_SCOPE_AGENT) < mygen) {
      __builtin_amdgcn_s_sleep(1);
    }
  }
  asm volatile("s_barrier_wait -3" ::: "memory");      // cluster barrier wait
  __syncthreads();
  __builtin_amdgcn_fence(__ATOMIC_ACQUIRE, "agent");   // invalidate stale lines
}

// ---- persistent scan kernel ------------------------------------------------
__global__ __launch_bounds__(kWaves * 32)
void k_gru_scan(const float* __restrict__ bzp, const float* __restrict__ bfp,
                const float* __restrict__ brp, float* __restrict__ out,
                unsigned char* __restrict__ ws) {
  // Static LDS: W tiles for this WG's 64 columns (4 waves x 3 gates x 8 kt).
  __shared__ __align__(16) unsigned short WL[kWaves][3][kKTw][32][16];  // 96 KB
  // Dynamic LDS: Uz|Uf packed slices for this WG, 2 x 128 KB, TDM-staged.
  extern __shared__ __align__(16) unsigned short UL[];

  const int tid  = threadIdx.x;
  const int wave = tid >> 5, lane = tid & 31;
  const int wg = blockIdx.x, group = wg >> 4, wgc = wg & 15;
  const int ntl = (wgc << 2) + wave;       // global N-tile 0..63
  const int hi  = lane >> 4, lc = lane & 15;
  const int col = (ntl << 4) + lc;         // this lane's output column
  const int bA  = (group << 4) + lc;       // batch row for A-operand assembly
  const int rowb = (group << 4) + (hi << 3);  // C-tile row base for this lane

  const unsigned short* ubf = (const unsigned short*)(ws + OFF_UBF);
  const unsigned short* Wpk = (const unsigned short*)(ws + OFF_WPK);
  const unsigned short* Upk = (const unsigned short*)(ws + OFF_UPK);
  float*          xf  = (float*)(ws + OFF_XF);
  unsigned short* xbf = (unsigned short*)(ws + OFF_XBF);
  unsigned short* gbf = (unsigned short*)(ws + OFF_GBF);
  unsigned* bar = (unsigned*)(ws + OFF_BAR);
  unsigned* cnt = bar + group * 32;   // 128B apart per group
  unsigned* gen = cnt + 16;

  // ---- TDM: stage this WG's Uz and Uf slices (2 x 128 KB, contiguous) ------
  {
    unsigned lds_off = (unsigned)(unsigned long long)(lds_us_ptr)UL;
    const unsigned short* gz = Upk + (size_t)(0 * kNT + (wgc << 2)) * kKTu * 32 * 16;
    const unsigned short* gf = Upk + (size_t)(1 * kNT + (wgc << 2)) * kKTu * 32 * 16;
    if (tid == 0) {
      tdm_load_1d(lds_off,            gz, 32768);   // 128 KB of Uz tiles
      tdm_load_1d(lds_off + 131072u,  gf, 32768);   // 128 KB of Uf tiles
    }
    __builtin_amdgcn_s_wait_tensorcnt(0);
  }

  // Stage this WG's W slices into LDS (3 gates x 8 kt x 32B per lane).
  for (int g = 0; g < 3; ++g) {
    for (int kt = 0; kt < kKTw; ++kt) {
      const unsigned short* src =
          Wpk + ((((size_t)g * kNT + ntl) * kKTw + kt) * 32 + lane) * 16;
      *(v8bf*)&WL[wave][g][kt][lane][0] = *(const v8bf*)src;
      *(v8bf*)&WL[wave][g][kt][lane][8] = *(const v8bf*)(src + 8);
    }
  }
  __syncthreads();

  const float biasz = bzp[col], biasfv = bfp[col], biasr = brp[col];
  // LDS views: UL = [gate(2)][wave(4)][kt(32)][lane(32)][16] bf16.
  const unsigned short* ULz = UL + (size_t)wave * kKTu * 32 * 16;
  const unsigned short* ULf = ULz + 65536;   // +128 KB in elements
  const unsigned short* UrT = Upk + (size_t)(2 * kNT + ntl) * kKTu * 32 * 16;

  unsigned mygen = 0;

  for (int t = 0; t < kT; ++t) {
    const unsigned short* uRow = ubf + ((size_t)bA * kT + t) * kNU + (hi << 3);

    // Register-cache the 8 u A-tiles; reused by phase A (z,f) and phase B (r).
    v16bf au[kKTw];
    #pragma unroll
    for (int kt = 0; kt < kKTw; ++kt) au[kt] = ld_a(uRow + (kt << 5));

    // ---------------- phase A : z and f gates -------------------------------
    v8f accz = {}; v8f accfv = {};
    #pragma unroll
    for (int kt = 0; kt < kKTw; ++kt) {          // input projection, K=256
      v16bf b0 = ld_b(&WL[wave][0][kt][lane][0]);
      v16bf b1 = ld_b(&WL[wave][1][kt][lane][0]);
      accz  = wmma_bf16(au[kt], b0, accz);
      accfv = wmma_bf16(au[kt], b1, accfv);
    }
    const unsigned short* xRow = xbf + ((size_t)bA << 10) + (hi << 3);
    #pragma unroll 8
    for (int kt = 0; kt < kKTu; ++kt) {          // recurrent, K=1024 (LDS B)
      v16bf a  = ld_a(xRow + (kt << 5));
      v16bf b0 = ld_b(ULz + (((kt << 5) + lane) << 4));
      v16bf b1 = ld_b(ULf + (((kt << 5) + lane) << 4));
      accz  = wmma_bf16(a, b0, accz);
      accfv = wmma_bf16(a, b1, accfv);
    }
    float zr[8], xr[8];
    #pragma unroll
    for (int i = 0; i < 8; ++i) {
      const size_t bi = (size_t)(rowb + i);
      float xv = xf[(bi << 10) + col];
      float zv = fsigmoid(accz[i]  + biasz);
      float fv = fsigmoid(accfv[i] + biasfv);
      gbf[(bi << 10) + col] = f2bf(fv * xv);     // g = f (.) x for the r-GEMM
      out[(bi * kT + t) * kNX + col] = xv;       // io_delay: ys[t] = x_t
      zr[i] = zv; xr[i] = xv;
    }
    cluster_sync(cnt, gen, mygen, tid);          // g visible cluster-wide

    // ---------------- phase B : r gate + state update -----------------------
    v8f accr = {};
    #pragma unroll
    for (int kt = 0; kt < kKTw; ++kt)
      accr = wmma_bf16(au[kt], ld_b(&WL[wave][2][kt][lane][0]), accr);
    const unsigned short* gRow = gbf + ((size_t)bA << 10) + (hi << 3);
    #pragma unroll 8
    for (int kt = 0; kt < kKTu; ++kt)            // recurrent, K=1024 (L2 B)
      accr = wmma_bf16(ld_a(gRow + (kt << 5)),
                       ld_b(UrT + (((kt << 5) + lane) << 4)), accr);
    #pragma unroll
    for (int i = 0; i < 8; ++i) {
      float rv = ftanh(accr[i] + biasr);
      float xn = zr[i] * xr[i] + (1.0f - zr[i]) * rv;
      const size_t bi = (size_t)(rowb + i);
      xf [(bi << 10) + col] = xn;
      xbf[(bi << 10) + col] = f2bf(xn);
    }
    cluster_sync(cnt, gen, mygen, tid);          // x_{t+1} visible cluster-wide
  }
}

// ---------------------------------------------------------------------------
extern "C" void kernel_launch(void* const* d_in, const int* in_sizes, int n_in,
                              void* d_out, int out_size, void* d_ws, size_t ws_size,
                              hipStream_t stream) {
  const float* u  = (const float*)d_in[0];
  const float* x0 = (const float*)d_in[1];
  const float* Wz = (const float*)d_in[2];
  const float* Uz = (const float*)d_in[3];
  const float* bz = (const float*)d_in[4];
  const float* Wf = (const float*)d_in[5];
  const float* Uf = (const float*)d_in[6];
  const float* bfv= (const float*)d_in[7];
  const float* Wr = (const float*)d_in[8];
  const float* Ur = (const float*)d_in[9];
  const float* br = (const float*)d_in[10];
  float* out = (float*)d_out;
  unsigned char* ws = (unsigned char*)d_ws;

  (void)in_sizes; (void)n_in; (void)out_size; (void)ws_size;

  k_convert_u<<<(kB * kT * kNU + 255) / 256, 256, 0, stream>>>(
      u, (unsigned short*)(ws + OFF_UBF), kB * kT * kNU);
  k_pack_w<<<(3 * kNT * kKTw * 32 + 255) / 256, 256, 0, stream>>>(
      Wz, Wf, Wr, (unsigned short*)(ws + OFF_WPK));
  k_pack_u<<<(3 * kNT * kKTu * 32 + 255) / 256, 256, 0, stream>>>(
      Uz, Uf, Ur, (unsigned short*)(ws + OFF_UPK));
  k_init_state<<<(kB * kNX + 255) / 256, 256, 0, stream>>>(
      x0, (float*)(ws + OFF_XF), (unsigned short*)(ws + OFF_XBF),
      (unsigned*)(ws + OFF_BAR));
  // 256 KB dynamic LDS for the TDM-staged Uz/Uf slices.
  k_gru_scan<<<kGroups * kWGsPerGp, kWaves * 32, 262144, stream>>>(
      bz, bfv, br, out, ws);
}
